// MultiHeadAttention_20126216749084
// MI455X (gfx1250) — matrix operand order repaired and checked by reading, not compiled, not run on hardware
//
#include <hip/hip_runtime.h>

#define D_MODEL  2048
#define NUM_HEAD 16
#define HEAD_DIM 128
#define SEQ      2048
#define BATCH    4

typedef __attribute__((ext_vector_type(16))) __bf16 v16bf;
typedef __attribute__((ext_vector_type(8)))  float  v8f;

struct alignas(16) I4 { int x, y, z, w; };
struct alignas(16) F4 { float x, y, z, w; };
struct alignas(8)  U2 { unsigned int x, y; };

union FragU { I4 q[2]; v16bf v; };

__device__ __forceinline__ unsigned short f2bf(float f) {
  unsigned int u = __float_as_uint(f);
  u += 0x7FFFu + ((u >> 16) & 1u);   // round-to-nearest-even
  return (unsigned short)(u >> 16);
}

// Two 16B LDS reads -> one 16x32 bf16 A fragment (K pairs split 8/16 by lane-half)
__device__ __forceinline__ v16bf ld_frag2(const unsigned short* p0, const unsigned short* p1) {
  FragU u;
  u.q[0] = *reinterpret_cast<const I4*>(p0);
  u.q[1] = *reinterpret_cast<const I4*>(p1);
  return u.v;
}
// 32 contiguous bytes -> B fragment (K = 16*half .. +15 contiguous per lane)
__device__ __forceinline__ v16bf ld_frag(const unsigned short* p) {
  return ld_frag2(p, p + 8);
}

__device__ __forceinline__ v8f wmma_bf16(v16bf a, v16bf b, v8f c) {
  return __builtin_amdgcn_wmma_f32_16x16x32_bf16(false, a, false, b, (short)0, c, false, false);
}

// CDNA5 async global->LDS copy, 16B per lane, tracked by ASYNCcnt.
// LDS destination VGPR holds the wave-relative LDS byte address = low 32 bits
// of the generic pointer to a __shared__ object.
__device__ __forceinline__ void async_g2l_b128(void* lds_dst, const void* gsrc) {
  unsigned loff = (unsigned)(unsigned long long)lds_dst;
  asm volatile("global_load_async_to_lds_b128 %0, %1, off"
               :: "v"(loff), "v"(gsrc)
               : "memory");
}
__device__ __forceinline__ void wait_async0() {
  asm volatile("s_wait_asynccnt 0x0" ::: "memory");
}

// ---------------------------------------------------------------------------
// Kernel 1: f32 -> bf16 conversion (vectorized 4-wide)
// ---------------------------------------------------------------------------
__global__ void __launch_bounds__(256)
cvt_kernel(const float* __restrict__ src, unsigned short* __restrict__ dst, int n4) {
  int i = blockIdx.x * blockDim.x + threadIdx.x;
  int stride = gridDim.x * blockDim.x;
  for (; i < n4; i += stride) {
    F4 f = reinterpret_cast<const F4*>(src)[i];
    U2 r;
    r.x = (unsigned int)f2bf(f.x) | ((unsigned int)f2bf(f.y) << 16);
    r.y = (unsigned int)f2bf(f.z) | ((unsigned int)f2bf(f.w) << 16);
    reinterpret_cast<U2*>(dst)[i] = r;
  }
}

// ---------------------------------------------------------------------------
// Kernel 2: QKV projection  qkv[m,o] = sum_k X[m,k] * Wp[o,k]
//   M = 8192 (b*s), N = 6144, K = 2048. 128x128 tile / block, 8 waves,
//   wave = 32x64 sub-tile (2x4 WMMA accumulators). Double-buffered LDS with
//   async global->LDS staging overlapping the WMMA block. Scatter epilogue
//   writes Q,K as [b,h,s,128] bf16 and V transposed [b,h,128,s] bf16.
// ---------------------------------------------------------------------------
__global__ void __launch_bounds__(256)
qkv_gemm_kernel(const unsigned short* __restrict__ X,
                const unsigned short* __restrict__ W,
                unsigned short* __restrict__ Qb,
                unsigned short* __restrict__ Kb,
                unsigned short* __restrict__ Vb) {
  __shared__ alignas(32) unsigned short As[2][128 * 32];
  __shared__ alignas(32) unsigned short Bs[2][128 * 32];
  const int m0 = blockIdx.x * 128;
  const int n0 = blockIdx.y * 128;
  const int tid  = threadIdx.x;
  const int lane = tid & 31;
  const int wave = tid >> 5;
  const int wm = wave & 3;   // 32-row block
  const int wn = wave >> 2;  // 64-col block
  const int ln = lane & 15;
  const int hf = lane >> 4;

  auto stage = [&](int buf, int kbase) {
#pragma unroll
    for (int it = 0; it < 2; ++it) {
      int idx = tid + it * 256;          // 0..511
      int r = idx >> 2;                  // 0..127
      int c = (idx & 3) << 3;            // 0,8,16,24 (bf16 elems)
      async_g2l_b128(&As[buf][r * 32 + c],
                     &X[(size_t)(m0 + r) * D_MODEL + kbase + c]);
      async_g2l_b128(&Bs[buf][r * 32 + c],
                     &W[(size_t)(n0 + r) * D_MODEL + kbase + c]);
    }
  };

  const v8f vzero = {};
  v8f acc[2][4];
#pragma unroll
  for (int i = 0; i < 2; ++i)
#pragma unroll
    for (int j = 0; j < 4; ++j) acc[i][j] = vzero;

  stage(0, 0);
  wait_async0();
  __syncthreads();

  for (int kb = 0; kb < D_MODEL; kb += 32) {
    const int buf = (kb >> 5) & 1;
    if (kb + 32 < D_MODEL) stage(buf ^ 1, kb + 32);   // prefetch next tile (async)

    v16bf af[2], bfv[4];
#pragma unroll
    for (int mt = 0; mt < 2; ++mt) {
      const unsigned short* pa = &As[buf][(wm * 32 + mt * 16 + ln) * 32];
      af[mt] = ld_frag2(pa + 8 * hf, pa + 16 + 8 * hf);
    }
#pragma unroll
    for (int nt = 0; nt < 4; ++nt)
      bfv[nt] = ld_frag2(&Bs[buf][(wn * 64 + nt * 16 + ln) * 32 + 8 * hf], &Bs[buf][(wn * 64 + nt * 16 + ln) * 32 + 16 + 8 * hf]);
#pragma unroll
    for (int mt = 0; mt < 2; ++mt)
#pragma unroll
      for (int nt = 0; nt < 4; ++nt)
        acc[mt][nt] = wmma_bf16(af[mt], bfv[nt], acc[mt][nt]);

    wait_async0();
    __syncthreads();
  }

  // Epilogue: C layout row = vgpr_j + 8*half, col = lane%16
#pragma unroll
  for (int mt = 0; mt < 2; ++mt) {
#pragma unroll
    for (int nt = 0; nt < 4; ++nt) {
#pragma unroll
      for (int j = 0; j < 8; ++j) {
        int row = m0 + wm * 32 + mt * 16 + hf * 8 + j;   // token index
        int o   = n0 + wn * 64 + nt * 16 + ln;           // output feature
        int bb = row >> 11;         // batch
        int ss = row & 2047;        // seq
        int which = o >> 11;        // 0=Q 1=K 2=V
        int rem = o & 2047;
        int hh = rem >> 7;
        int dd = rem & 127;
        unsigned short val = f2bf(acc[mt][nt][j]);
        size_t bh = (size_t)(bb * NUM_HEAD + hh);
        if (which == 0)      Qb[(bh * SEQ + ss) * HEAD_DIM + dd] = val;
        else if (which == 1) Kb[(bh * SEQ + ss) * HEAD_DIM + dd] = val;
        else                 Vb[(bh * HEAD_DIM + dd) * SEQ + ss] = val; // V transposed
      }
    }
  }
}

// ---------------------------------------------------------------------------
// Kernel 3: causal flash attention.
//   Block = (128 q-rows, one head, one batch); 8 waves, 16 q-rows each.
//   KV tiles of 64 staged in LDS via async-to-LDS (K row-major [64][128],
//   V^T [128][64]). Online softmax; P restriped C->A layout via per-wave LDS.
// ---------------------------------------------------------------------------
__global__ void __launch_bounds__(256)
flash_attn_kernel(const unsigned short* __restrict__ Qb,
                  const unsigned short* __restrict__ Kb,
                  const unsigned short* __restrict__ Vb,
                  unsigned short* __restrict__ Ob) {
  __shared__ alignas(32) unsigned short Ksh[64 * 128];   // [kv][d]
  __shared__ alignas(32) unsigned short Vsh[128 * 64];   // [d][kv]
  __shared__ alignas(32) unsigned short Psh[8][16 * 64]; // per-wave P tile

  const int q0 = blockIdx.x * 128;
  const int h  = blockIdx.y;
  const int b  = blockIdx.z;
  const size_t bh = (size_t)(b * NUM_HEAD + h);
  const unsigned short* Qh = Qb + bh * SEQ * HEAD_DIM;
  const unsigned short* Kh = Kb + bh * SEQ * HEAD_DIM;
  const unsigned short* Vh = Vb + bh * HEAD_DIM * SEQ;   // [d][s]

  const int tid  = threadIdx.x;
  const int lane = tid & 31;
  const int wave = tid >> 5;
  const int ln = lane & 15;
  const int hf = lane >> 4;
  const int qr0 = q0 + wave * 16;

  // Q fragments: 16x128 as 4 x (16x32), loaded straight from global (A layout)
  v16bf qf[4];
#pragma unroll
  for (int kc = 0; kc < 4; ++kc) {
    const unsigned short* p = &Qh[(size_t)(qr0 + ln) * HEAD_DIM + kc * 32 + 8 * hf];
    qf[kc] = ld_frag2(p, p + 16);
  }

  const v8f vzero = {};
  v8f O[8];
  float m_i[8], l_i[8];
#pragma unroll
  for (int nt = 0; nt < 8; ++nt) O[nt] = vzero;
#pragma unroll
  for (int j = 0; j < 8; ++j) { m_i[j] = -1e30f; l_i[j] = 0.0f; }

  const float scale = 0.08838834764831845f;  // 1/sqrt(128)
  const int kv_end = q0 + 128;               // causal upper bound for this block

  for (int kv = 0; kv < kv_end; kv += 64) {
    // Stage K tile [64][128] and V^T tile [128][64] via async global->LDS
#pragma unroll
    for (int it = 0; it < 4; ++it) {
      int idx = tid + it * 256;              // 0..1023
      int r  = idx >> 4;                     // 0..63
      int c  = (idx & 15) << 3;              // 0..120
      async_g2l_b128(&Ksh[r * 128 + c], &Kh[(size_t)(kv + r) * HEAD_DIM + c]);
      int dr = idx >> 3;                     // 0..127
      int dc = (idx & 7) << 3;               // 0..56
      async_g2l_b128(&Vsh[dr * 64 + dc], &Vh[(size_t)dr * SEQ + kv + dc]);
    }
    wait_async0();
    __syncthreads();

    if (kv <= qr0 + 15) {   // uniform per wave: skip fully-masked tiles
      // S = scale * Q K^T, with causal mask. 4 column tiles of 16.
      v8f Sv[4];
#pragma unroll
      for (int nt = 0; nt < 4; ++nt) {
        v8f s = vzero;
#pragma unroll
        for (int kc = 0; kc < 4; ++kc) {
          v16bf kb16 = ld_frag2(&Ksh[(nt * 16 + ln) * 128 + kc * 32 + 8 * hf], &Ksh[(nt * 16 + ln) * 128 + kc * 32 + 16 + 8 * hf]);
          s = wmma_bf16(qf[kc], kb16, s);
        }
        int col = kv + nt * 16 + ln;
#pragma unroll
        for (int j = 0; j < 8; ++j) {
          int row = qr0 + hf * 8 + j;
          s[j] = (col <= row) ? s[j] * scale : -1e30f;
        }
        Sv[nt] = s;
      }

      // row max across 4 tiles + 16 lanes of this half
      float mnew[8];
#pragma unroll
      for (int j = 0; j < 8; ++j) {
        float mx = fmaxf(fmaxf(Sv[0][j], Sv[1][j]), fmaxf(Sv[2][j], Sv[3][j]));
#pragma unroll
        for (int off = 8; off >= 1; off >>= 1)
          mx = fmaxf(mx, __shfl_xor(mx, off, 32));
        mnew[j] = fmaxf(m_i[j], mx);
      }

      // P = exp(S - mnew); write bf16 to per-wave LDS (C layout -> memory)
      float rsum[8];
#pragma unroll
      for (int j = 0; j < 8; ++j) rsum[j] = 0.0f;
#pragma unroll
      for (int nt = 0; nt < 4; ++nt) {
#pragma unroll
        for (int j = 0; j < 8; ++j) {
          float p = __expf(Sv[nt][j] - mnew[j]);
          rsum[j] += p;
          Psh[wave][(hf * 8 + j) * 64 + nt * 16 + ln] = f2bf(p);
        }
      }
#pragma unroll
      for (int j = 0; j < 8; ++j) {
#pragma unroll
        for (int off = 8; off >= 1; off >>= 1)
          rsum[j] += __shfl_xor(rsum[j], off, 32);
      }

      // rescale running state
#pragma unroll
      for (int j = 0; j < 8; ++j) {
        float alpha = __expf(m_i[j] - mnew[j]);
        m_i[j] = mnew[j];
        l_i[j] = l_i[j] * alpha + rsum[j];
#pragma unroll
        for (int nt = 0; nt < 8; ++nt) O[nt][j] *= alpha;
      }

      // O += P @ V : P 16x64 as two A fragments, V^T gives contiguous B frags
      v16bf pf[2];
#pragma unroll
      for (int kc = 0; kc < 2; ++kc) {
        const unsigned short* pp = &Psh[wave][ln * 64 + kc * 32 + 8 * hf];
        pf[kc] = ld_frag2(pp, pp + 16);
      }
#pragma unroll
      for (int nt = 0; nt < 8; ++nt) {
#pragma unroll
        for (int kc = 0; kc < 2; ++kc) {
          v16bf vb16 = ld_frag2(&Vsh[(nt * 16 + ln) * 64 + kc * 32 + 8 * hf], &Vsh[(nt * 16 + ln) * 64 + kc * 32 + 16 + 8 * hf]);
          O[nt] = wmma_bf16(pf[kc], vb16, O[nt]);
        }
      }
    }
    __syncthreads();
  }

  // epilogue: O /= l, merged-head layout [b*s][d_model] bf16
#pragma unroll
  for (int j = 0; j < 8; ++j) l_i[j] = 1.0f / l_i[j];
#pragma unroll
  for (int nt = 0; nt < 8; ++nt) {
#pragma unroll
    for (int j = 0; j < 8; ++j) {
      int row = qr0 + hf * 8 + j;
      int col = h * HEAD_DIM + nt * 16 + ln;
      Ob[((size_t)(b * SEQ + row)) * D_MODEL + col] = f2bf(O[nt][j] * l_i[j]);
    }
  }
}

// ---------------------------------------------------------------------------
// Kernel 4: output projection  out[m,o] = sum_k A[m,k] * Wo[o,k], f32 out.
//   Same double-buffered async-staged structure as the QKV GEMM.
// ---------------------------------------------------------------------------
__global__ void __launch_bounds__(256)
out_gemm_kernel(const unsigned short* __restrict__ A,
                const unsigned short* __restrict__ W,
                float* __restrict__ Out) {
  __shared__ alignas(32) unsigned short As[2][128 * 32];
  __shared__ alignas(32) unsigned short Bs[2][128 * 32];
  const int m0 = blockIdx.x * 128;
  const int n0 = blockIdx.y * 128;
  const int tid  = threadIdx.x;
  const int lane = tid & 31;
  const int wave = tid >> 5;
  const int wm = wave & 3;
  const int wn = wave >> 2;
  const int ln = lane & 15;
  const int hf = lane >> 4;

  auto stage = [&](int buf, int kbase) {
#pragma unroll
    for (int it = 0; it < 2; ++it) {
      int idx = tid + it * 256;
      int r = idx >> 2;
      int c = (idx & 3) << 3;
      async_g2l_b128(&As[buf][r * 32 + c],
                     &A[(size_t)(m0 + r) * D_MODEL + kbase + c]);
      async_g2l_b128(&Bs[buf][r * 32 + c],
                     &W[(size_t)(n0 + r) * D_MODEL + kbase + c]);
    }
  };

  const v8f vzero = {};
  v8f acc[2][4];
#pragma unroll
  for (int i = 0; i < 2; ++i)
#pragma unroll
    for (int j = 0; j < 4; ++j) acc[i][j] = vzero;

  stage(0, 0);
  wait_async0();
  __syncthreads();

  for (int kb = 0; kb < D_MODEL; kb += 32) {
    const int buf = (kb >> 5) & 1;
    if (kb + 32 < D_MODEL) stage(buf ^ 1, kb + 32);

    v16bf af[2], bfv[4];
#pragma unroll
    for (int mt = 0; mt < 2; ++mt) {
      const unsigned short* pa = &As[buf][(wm * 32 + mt * 16 + ln) * 32];
      af[mt] = ld_frag2(pa + 8 * hf, pa + 16 + 8 * hf);
    }
#pragma unroll
    for (int nt = 0; nt < 4; ++nt)
      bfv[nt] = ld_frag2(&Bs[buf][(wn * 64 + nt * 16 + ln) * 32 + 8 * hf], &Bs[buf][(wn * 64 + nt * 16 + ln) * 32 + 16 + 8 * hf]);
#pragma unroll
    for (int mt = 0; mt < 2; ++mt)
#pragma unroll
      for (int nt = 0; nt < 4; ++nt)
        acc[mt][nt] = wmma_bf16(af[mt], bfv[nt], acc[mt][nt]);

    wait_async0();
    __syncthreads();
  }

#pragma unroll
  for (int mt = 0; mt < 2; ++mt)
#pragma unroll
    for (int nt = 0; nt < 4; ++nt)
#pragma unroll
      for (int j = 0; j < 8; ++j) {
        int row = m0 + wm * 32 + mt * 16 + hf * 8 + j;
        int col = n0 + wn * 64 + nt * 16 + ln;
        Out[(size_t)row * D_MODEL + col] = acc[mt][nt][j];
      }
}

// ---------------------------------------------------------------------------
extern "C" void kernel_launch(void* const* d_in, const int* in_sizes, int n_in,
                              void* d_out, int out_size, void* d_ws, size_t ws_size,
                              hipStream_t stream) {
  (void)in_sizes; (void)n_in; (void)out_size; (void)ws_size;
  const float* x  = (const float*)d_in[0];   // [4,2048,2048]
  const float* wp = (const float*)d_in[1];   // [6144,2048]
  const float* wo = (const float*)d_in[2];   // [2048,2048]
  float* out = (float*)d_out;                // [4,2048,2048]

  // workspace partition (bf16 buffers), ~201 MB total
  unsigned short* xb  = (unsigned short*)d_ws;                       // [8192][2048]
  unsigned short* wpb = xb  + (size_t)BATCH * SEQ * D_MODEL;         // [6144][2048]
  unsigned short* wob = wpb + (size_t)3 * D_MODEL * D_MODEL;         // [2048][2048]
  unsigned short* Qb  = wob + (size_t)D_MODEL * D_MODEL;             // [b,h,s,128]
  const size_t per = (size_t)BATCH * NUM_HEAD * SEQ * HEAD_DIM;
  unsigned short* Kb = Qb + per;                                     // [b,h,s,128]
  unsigned short* Vb = Kb + per;                                     // [b,h,128,s]
  unsigned short* Ab = Vb + per;                                     // [8192][2048]

  cvt_kernel<<<dim3(1024), dim3(256), 0, stream>>>(x,  xb,  (BATCH * SEQ * D_MODEL) / 4);
  cvt_kernel<<<dim3(1024), dim3(256), 0, stream>>>(wp, wpb, (3 * D_MODEL * D_MODEL) / 4);
  cvt_kernel<<<dim3(1024), dim3(256), 0, stream>>>(wo, wob, (D_MODEL * D_MODEL) / 4);

  qkv_gemm_kernel<<<dim3(64, 48), dim3(256), 0, stream>>>(xb, wpb, Qb, Kb, Vb);
  flash_attn_kernel<<<dim3(SEQ / 128, NUM_HEAD, BATCH), dim3(256), 0, stream>>>(Qb, Kb, Vb, Ab);
  out_gemm_kernel<<<dim3(64, 16), dim3(256), 0, stream>>>(Ab, wob, out);
}
